// ScaledDotProductAttention_66872640799209
// MI455X (gfx1250) — matrix operand order repaired and checked by reading, not compiled, not run on hardware
//
#include <hip/hip_runtime.h>
#include <hip/hip_bf16.h>

// ---------------------------------------------------------------------------
// ScaledDotProductAttention (non-standard: softmax over the QUERY axis)
//   B=4, S=4096, D=64, f32 in / f32 out.
// Strategy (MI455X / gfx1250, wave32, WMMA f16->f32):
//   K1: column stats  m_k, 1/Z_k  of softmax(Q.K^T/64, axis=q)   (1 GEMM sweep)
//   K2: attn[q,d] = sum_k exp(s[q,k]-m_k)/Z_k * V[k,d]           (2 GEMM sweeps,
//       scores recomputed TRANSPOSED so the f32 C-layout == f16 A-layout of P)
//   K3: attn_w = softmax(attn, axis=q)                            (tiny)
// Everything is L2-resident (12 MB inputs vs 192 MB L2); compute-bound on the
// WMMA pipe, hence f16 operands with f32 accumulation.
// ---------------------------------------------------------------------------

typedef __attribute__((ext_vector_type(16))) _Float16 v16h;
typedef __attribute__((ext_vector_type(8)))  float    v8f;

constexpr int BATCH = 4;
constexpr int S     = 4096;
constexpr int D     = 64;
constexpr float SCALE = 1.0f / 64.0f;   // reference divides by k.shape[-1]

__device__ __forceinline__ v8f wmma_f16(v16h a, v16h b, v8f c) {
    // v_wmma_f32_16x16x32_f16: (neg_a, A, neg_b, B, c_mod, C, reuse_a, reuse_b)
    return __builtin_amdgcn_wmma_f32_16x16x32_f16(false, a, false, b,
                                                  (short)0, c, false, false);
}

// 16 contiguous f32 -> v16h (contiguous k order, not AMD's table order: lane holds one column,
// contraction indices d0..d0+15 contiguous)
__device__ __forceinline__ v16h load_cvt16(const float* __restrict__ p) {
    v16h r;
#pragma unroll
    for (int i = 0; i < 16; ++i) r[i] = (_Float16)p[i];
    return r;
}

// two 8-f32 chunks -> v16h (A-operand layout: lane holds one row,
// contraction {base..base+7} in elems 0..7 and {base+16..base+23} in 8..15)
__device__ __forceinline__ v16h load_cvt8_8(const float* __restrict__ p0,
                                            const float* __restrict__ p1) {
    v16h r;
#pragma unroll
    for (int i = 0; i < 8; ++i) { r[i] = (_Float16)p0[i]; r[i + 8] = (_Float16)p1[i]; }
    return r;
}

// ---------------------------------------------------------------------------
// Kernel 1: per-key-column online softmax stats over the query axis.
// grid = BATCH*(S/64), block = 128 (4 waves); each wave owns 16 key columns.
// S-tile orientation: A = Q rows (16q x 32d), B = K^T (32d x 16k) -> C lane
// holds ONE key column and 8 query rows => column reduce = in-lane + xor16.
// ---------------------------------------------------------------------------
__global__ void __launch_bounds__(128)
sdpa_col_stats_kernel(const float* __restrict__ q, const float* __restrict__ k,
                      float* __restrict__ colmax, float* __restrict__ colrcp) {
    const int wg    = blockIdx.x;
    const int b     = wg / (S / 64);
    const int kblk  = wg % (S / 64);
    const int lane  = threadIdx.x & 31;
    const int wave  = threadIdx.x >> 5;
    const int kBase = kblk * 64 + wave * 16;
    const int l15   = lane & 15;
    const bool hi   = lane >= 16;

    const float* Qb = q + (size_t)b * S * D;
    const float* Kb = k + (size_t)b * S * D;

    // Hoisted B operand (K^T): lane = key column kBase+l15.
    // chunk d0: lo lanes hold d {d0+0..7, d0+16..23}, hi lanes +8.
    const float* kp = Kb + (size_t)(kBase + l15) * D + (hi ? 8 : 0);
    const v16h bK0 = load_cvt8_8(kp,      kp + 16);   // d = 0..31
    const v16h bK1 = load_cvt8_8(kp + 32, kp + 48);   // d = 32..63

    float m = -__builtin_inff();
    float Z = 0.0f;

    for (int qt = 0; qt < S; qt += 16) {
        // A operand (Q rows): lo lanes d {d0+0..7, d0+16..23}; hi lanes +8.
        const float* qp = Qb + (size_t)(qt + l15) * D + (hi ? 8 : 0);
        const v16h a0 = load_cvt8_8(qp,      qp + 16);
        const v16h a1 = load_cvt8_8(qp + 32, qp + 48);
        v8f c = {};
        c = wmma_f16(a0, bK0, c);
        c = wmma_f16(a1, bK1, c);   // accumulate second d-chunk

        // c[r] = score(q = qt + r + (hi?8:0), key = kBase+l15) (unscaled)
        float tmax = -__builtin_inff();
#pragma unroll
        for (int r = 0; r < 8; ++r) { float s = c[r] * SCALE; c[r] = s; tmax = fmaxf(tmax, s); }
        tmax = fmaxf(tmax, __shfl_xor(tmax, 16, 32));   // join rows 0-7 with 8-15
        const float mnew = fmaxf(m, tmax);
        float tsum = 0.0f;
#pragma unroll
        for (int r = 0; r < 8; ++r) tsum += __expf(c[r] - mnew);
        tsum += __shfl_xor(tsum, 16, 32);
        Z = Z * __expf(m - mnew) + tsum;    // exp(-inf - x) == 0 on first iter
        m = mnew;
    }

    if (!hi) {                               // lo/hi lanes hold identical stats
        colmax[b * S + kBase + l15] = m;
        colrcp[b * S + kBase + l15] = 1.0f / Z;   // Z >= 1 (max term is exp(0))
    }
}

// ---------------------------------------------------------------------------
// Kernel 2: attn = P @ V with P recomputed TRANSPOSED.
// grid = BATCH*(S/64), block = 128; each wave owns 16 query rows, streams all k.
// S^T tile (A = K rows, B = Q^T): the f32 C layout of two stacked 16x16 tiles
// (k = kt..kt+15 and kt+16..kt+31) is exactly the f16 A-operand layout of the
// 16q x 32k P tile -> pure in-register exp+cvt, no LDS transpose.
// ---------------------------------------------------------------------------
__global__ void __launch_bounds__(128)
sdpa_attn_kernel(const float* __restrict__ q, const float* __restrict__ k,
                 const float* __restrict__ v, const float* __restrict__ colmax,
                 const float* __restrict__ colrcp, float* __restrict__ attn) {
    const int wg    = blockIdx.x;
    const int b     = wg / (S / 64);
    const int qblk  = wg % (S / 64);
    const int lane  = threadIdx.x & 31;
    const int wave  = threadIdx.x >> 5;
    const int qBase = qblk * 64 + wave * 16;
    const int l15   = lane & 15;
    const bool hi   = lane >= 16;

    const float* Qb = q + (size_t)b * S * D;
    const float* Kb = k + (size_t)b * S * D;
    const float* Vb = v + (size_t)b * S * D;
    const float* mB = colmax + b * S;
    const float* rB = colrcp + b * S;

    // Hoisted B operand (Q^T): lane = query column qBase+l15.
    const float* qp = Qb + (size_t)(qBase + l15) * D + (hi ? 8 : 0);
    const v16h bQ0 = load_cvt8_8(qp,      qp + 16);
    const v16h bQ1 = load_cvt8_8(qp + 32, qp + 48);

    v8f acc[4] = {v8f{}, v8f{}, v8f{}, v8f{}};   // 16q x 64d accumulator

    for (int kt = 0; kt < S; kt += 32) {
        // --- S^T tiles: T0 = K[kt..kt+15] . Q^T, T1 = K[kt+16..kt+31] . Q^T
        const float* kp0 = Kb + (size_t)(kt + l15) * D + (hi ? 8 : 0);
        const float* kp1 = kp0 + (size_t)16 * D;
        const v16h aK00 = load_cvt8_8(kp0,      kp0 + 16);
        const v16h aK01 = load_cvt8_8(kp0 + 32, kp0 + 48);
        const v16h aK10 = load_cvt8_8(kp1,      kp1 + 16);
        const v16h aK11 = load_cvt8_8(kp1 + 32, kp1 + 48);
        v8f t0 = {}, t1 = {};
        t0 = wmma_f16(aK00, bQ0, t0);
        t0 = wmma_f16(aK01, bQ1, t0);
        t1 = wmma_f16(aK10, bQ0, t1);
        t1 = wmma_f16(aK11, bQ1, t1);

        // --- P operand: exp((s - m_k)/..)*rcpZ_k, C layout -> A layout directly.
        // T0 rows: k = kt + r + (hi?8:0)  -> A elems 0..7  (contraction 0..7 / 8..15)
        // T1 rows: k = kt+16 + r + (hi?8:0) -> A elems 8..15 (contraction 16..23/24..31)
        const int kr0 = kt + (hi ? 8 : 0);
        const int kr1 = kr0 + 16;
        v16h aP;
#pragma unroll
        for (int r = 0; r < 8; ++r) {
            const float p0 = __expf(t0[r] * SCALE - mB[kr0 + r]) * rB[kr0 + r];
            const float p1 = __expf(t1[r] * SCALE - mB[kr1 + r]) * rB[kr1 + r];
            aP[r]     = (_Float16)p0;
            aP[r + 8] = (_Float16)p1;
        }

        // --- V B-operand per 16-wide d sub-tile: lane = col d, rows = k.
        const float* vp = Vb + (size_t)(kt + (hi ? 8 : 0)) * D + l15;
#pragma unroll
        for (int dt = 0; dt < 4; ++dt) {
            v16h vB;
#pragma unroll
            for (int r = 0; r < 16; ++r) vB[r] = (_Float16)vp[(size_t)(r + (r & 8)) * D + dt * 16];
            acc[dt] = wmma_f16(aP, vB, acc[dt]);
        }
    }

    // acc[dt]: rows q = qBase + r + (hi?8:0), col d = dt*16 + l15
    float* ob = attn + (size_t)b * S * D;
    const int qr = qBase + (hi ? 8 : 0);
#pragma unroll
    for (int dt = 0; dt < 4; ++dt)
#pragma unroll
        for (int r = 0; r < 8; ++r)
            ob[(size_t)(qr + r) * D + dt * 16 + l15] = acc[dt][r];
}

// ---------------------------------------------------------------------------
// Kernel 3: attn_w = softmax(attn, axis=q). One block per (b,d) column.
// ---------------------------------------------------------------------------
__global__ void __launch_bounds__(256)
sdpa_colsoftmax_kernel(const float* __restrict__ attn, float* __restrict__ out) {
    const int col = blockIdx.x;          // 0 .. BATCH*D-1
    const int b = col / D, d = col % D;
    const float* base = attn + (size_t)b * S * D + d;
    float* obase      = out  + (size_t)b * S * D + d;
    __shared__ float red[256];
    const int t = threadIdx.x;

    float m = -__builtin_inff();
    for (int i = t; i < S; i += 256) m = fmaxf(m, base[(size_t)i * D]);
    red[t] = m; __syncthreads();
    for (int o = 128; o > 0; o >>= 1) { if (t < o) red[t] = fmaxf(red[t], red[t + o]); __syncthreads(); }
    m = red[0]; __syncthreads();

    float s = 0.0f;
    for (int i = t; i < S; i += 256) s += __expf(base[(size_t)i * D] - m);
    red[t] = s; __syncthreads();
    for (int o = 128; o > 0; o >>= 1) { if (t < o) red[t] += red[t + o]; __syncthreads(); }
    const float rcp = 1.0f / red[0];

    for (int i = t; i < S; i += 256)
        obase[(size_t)i * D] = __expf(base[(size_t)i * D] - m) * rcp;
}

// ---------------------------------------------------------------------------
extern "C" void kernel_launch(void* const* d_in, const int* in_sizes, int n_in,
                              void* d_out, int out_size, void* d_ws, size_t ws_size,
                              hipStream_t stream) {
    const float* q = (const float*)d_in[0];
    const float* k = (const float*)d_in[1];
    const float* v = (const float*)d_in[2];

    float* attn   = (float*)d_out;                      // [B,S,D]
    float* attn_w = attn + (size_t)BATCH * S * D;       // [B,S,D]

    float* colmax = (float*)d_ws;                       // [B,S]
    float* colrcp = colmax + (size_t)BATCH * S;         // [B,S]  (128 KB total)

    sdpa_col_stats_kernel<<<BATCH * (S / 64), 128, 0, stream>>>(q, k, colmax, colrcp);
    sdpa_attn_kernel     <<<BATCH * (S / 64), 128, 0, stream>>>(q, k, v, colmax, colrcp, attn);
    sdpa_colsoftmax_kernel<<<BATCH * D, 256, 0, stream>>>(attn, attn_w);
}
